// AdvancedHistoryTransformer_65034394796606
// MI455X (gfx1250) — matrix operand order repaired and checked by reading, not compiled, not run on hardware
//
#include <hip/hip_runtime.h>
#include <hip/hip_bf16.h>
#include <math.h>

#define B_  256
#define S_  128
#define L_  40000
#define D_  128
#define DFF_ 256
#define NL_ 4
#define NH_ 8
#define TOPK_ 2500
#define VISW_ (L_/32)
#define PI_F 3.14159265358979f

typedef __attribute__((ext_vector_type(16))) _Float16 v16h;
typedef __attribute__((ext_vector_type(8)))  float    v8f;
typedef __attribute__((ext_vector_type(4)))  unsigned int v4u;
typedef __attribute__((ext_vector_type(8)))  int v8i_t;
typedef __attribute__((ext_vector_type(4)))  int v4i_t;

union V16 { uint4 u4[2]; _Float16 h[16]; v16h v; };

__device__ inline v8f wmma_f16(v16h a, v16h b, v8f c) {
  return __builtin_amdgcn_wmma_f32_16x16x32_f16(false, a, false, b, (short)0, c, false, false);
}
__device__ inline float geluf(float x) { return 0.5f * x * (1.0f + erff(x * 0.70710678118f)); }
__device__ inline float sigmoidf_(float x) { return 1.0f / (1.0f + expf(-x)); }

// ---------------- utility kernels ----------------
__global__ void f32_to_f16_kernel(const float* __restrict__ in, _Float16* __restrict__ out, int n) {
  int i = blockIdx.x * blockDim.x + threadIdx.x;
  if (i < n) out[i] = (_Float16)in[i];
}

__global__ void locpad_kernel(const float* __restrict__ loc_emb, _Float16* __restrict__ loc16) {
  int i = blockIdx.x * blockDim.x + threadIdx.x;
  if (i >= L_ * 64) return;
  int n = i >> 6, j = i & 63;
  loc16[i] = (j < 56) ? (_Float16)loc_emb[n * 56 + j] : (_Float16)0.0f;
}

__global__ void topk_scatter_kernel(const int* __restrict__ idx, int* __restrict__ pos) {
  int i = blockIdx.x * blockDim.x + threadIdx.x;
  if (i < TOPK_) pos[idx[i]] = i;
}

// ---------------- history ----------------
__global__ void hist_kernel(const int* __restrict__ loc_seq, const int* __restrict__ mask,
                            const float* __restrict__ hd, const float* __restrict__ rw,
                            float* __restrict__ hist, unsigned* __restrict__ visited,
                            int* __restrict__ vlen_ws) {
  int b = blockIdx.x, t = threadIdx.x; // t = s, 128 threads
  __shared__ int red[128];
  int mk = mask[b * S_ + t];
  red[t] = mk ? 1 : 0;
  __syncthreads();
  for (int st = 64; st > 0; st >>= 1) { if (t < st) red[t] += red[t + st]; __syncthreads(); }
  int vlen = red[0];
  if (t == 0) vlen_ws[b] = vlen;
  int loc = loc_seq[b * S_ + t];
  if (mk) {
    atomicOr(&visited[b * VISW_ + (loc >> 5)], 1u << (loc & 31));
    if (loc != 0) {
      float fv = (float)vlen;
      float rec = (t + 1.0f) / fmaxf(fv, 1.0f);
      float boost = sigmoidf_(rw[0] * (rec - 0.5f));
      float w = powf(hd[0], fv - (float)t - 1.0f) * (1.0f + boost);
      atomicAdd(&hist[(size_t)b * L_ + loc], w);
    }
  }
}

__global__ void histmax_kernel(const float* __restrict__ freq, float* __restrict__ hist,
                               float* __restrict__ rowmax) {
  int b = blockIdx.x, t = threadIdx.x; // 256 threads
  __shared__ float red[256];
  float mx = 0.0f;
  for (int i = t; i < L_; i += 256) {
    float fw = 1.0f / (logf(freq[i] + 1.0f) + 1.0f);
    float v = hist[(size_t)b * L_ + i] * fw;
    hist[(size_t)b * L_ + i] = v;
    mx = fmaxf(mx, v);
  }
  red[t] = mx;
  __syncthreads();
  for (int st = 128; st > 0; st >>= 1) { if (t < st) red[t] = fmaxf(red[t], red[t + st]); __syncthreads(); }
  if (t == 0) rowmax[b] = red[0];
}

// ---------------- embedding / temporal features / input LN + pos-enc ----------------
__global__ void embed_kernel(const int* __restrict__ loc_seq, const int* __restrict__ user_seq,
                             const int* __restrict__ weekday, const int* __restrict__ start_min,
                             const int* __restrict__ dur, const int* __restrict__ diff,
                             const float* __restrict__ loc_emb, const float* __restrict__ user_emb,
                             const float* __restrict__ tproj_w, const float* __restrict__ tproj_b,
                             const float* __restrict__ tln_g, const float* __restrict__ tln_b,
                             const float* __restrict__ in_ln_g, const float* __restrict__ in_ln_b,
                             float* __restrict__ x) {
  int tok = blockIdx.x;        // b*S + s
  int s = tok & (S_ - 1);
  int j = threadIdx.x;         // 128 threads
  __shared__ float red[128];
  __shared__ float temb[64];
  float hours = (float)start_min[tok] / 60.0f;
  float hr = hours / 24.0f * 2.0f * PI_F;
  float wd = (float)weekday[tok] / 7.0f * 2.0f * PI_F;
  int tcat = (int)(hours / 6.0f); tcat = tcat < 0 ? 0 : (tcat > 3 ? 3 : tcat);
  float f[10];
  f[0] = sinf(hr); f[1] = cosf(hr); f[2] = sinf(wd); f[3] = cosf(wd);
  f[4] = log1pf((float)dur[tok]) / 8.0f; f[5] = log1pf((float)diff[tok]) / 5.0f;
  f[6] = f[7] = f[8] = f[9] = 0.0f; f[6 + tcat] = 1.0f;
  float tp = 0.0f;
  if (j < 56) {
    for (int i = 0; i < 10; i++) tp += tproj_w[j * 10 + i] * f[i];
    tp += tproj_b[j];
  }
  red[j] = (j < 56) ? tp : 0.0f; __syncthreads();
  for (int st = 64; st > 0; st >>= 1) { if (j < st) red[j] += red[j + st]; __syncthreads(); }
  float mean = red[0] / 56.0f; __syncthreads();
  float d0 = (j < 56) ? (tp - mean) : 0.0f;
  red[j] = d0 * d0; __syncthreads();
  for (int st = 64; st > 0; st >>= 1) { if (j < st) red[j] += red[j + st]; __syncthreads(); }
  float var = red[0] / 56.0f; __syncthreads();
  if (j < 56) temb[j] = fmaxf(0.0f, (tp - mean) * rsqrtf(var + 1e-5f) * tln_g[j] + tln_b[j]);
  __syncthreads();
  int loc = loc_seq[tok], us = user_seq[tok];
  float c;
  if (j < 56)       c = loc_emb[(size_t)loc * 56 + j];
  else if (j < 72)  c = user_emb[(size_t)us * 16 + (j - 56)];
  else              c = temb[j - 72];
  red[j] = c; __syncthreads();
  for (int st = 64; st > 0; st >>= 1) { if (j < st) red[j] += red[j + st]; __syncthreads(); }
  float mean2 = red[0] / 128.0f; __syncthreads();
  float d1 = c - mean2;
  red[j] = d1 * d1; __syncthreads();
  for (int st = 64; st > 0; st >>= 1) { if (j < st) red[j] += red[j + st]; __syncthreads(); }
  float var2 = red[0] / 128.0f;
  float xv = d1 * rsqrtf(var2 + 1e-5f) * in_ln_g[j] + in_ln_b[j];
  int i2 = j >> 1;
  float div = expf(-(float)(2 * i2) * logf(10000.0f) / 128.0f);
  xv += ((j & 1) == 0) ? sinf((float)s * div) : cosf((float)s * div);
  x[(size_t)tok * D_ + j] = xv;
}

// ---------------- LayerNorm rows f32 -> padded f16 ----------------
__global__ void ln_kernel(const float* __restrict__ in, const float* __restrict__ g,
                          const float* __restrict__ b, _Float16* __restrict__ out,
                          int width, int padw) {
  int row = blockIdx.x, j = threadIdx.x;  // blockDim == padw (pow2)
  __shared__ float red[256];
  float v = (j < width) ? in[(size_t)row * width + j] : 0.0f;
  red[j] = v; __syncthreads();
  for (int st = blockDim.x >> 1; st > 0; st >>= 1) { if (j < st) red[j] += red[j + st]; __syncthreads(); }
  float mean = red[0] / (float)width; __syncthreads();
  float d = (j < width) ? (v - mean) : 0.0f;
  red[j] = d * d; __syncthreads();
  for (int st = blockDim.x >> 1; st > 0; st >>= 1) { if (j < st) red[j] += red[j + st]; __syncthreads(); }
  float var = red[0] / (float)width;
  float o = (j < width) ? (d * rsqrtf(var + 1e-5f) * g[j] + b[j]) : 0.0f;
  out[(size_t)row * padw + j] = (_Float16)o;
}

// ---------------- generic WMMA GEMM: out = A[M,K] @ W[N,K]^T + bias ----------------
// 1x4 register-blocked: one wave computes a 16x64 strip (4 accumulators).
// CONTRACT: M % 16 == 0, K % 32 == 0, and W has at least ceil(N/16)*16 readable
//           (zero-padded) rows so fragment loads need no predication.
// mode 0: store f16; 1: store f16 gelu; 2: f32 residual +=; 3: store f32
__global__ __launch_bounds__(128) void gemm16_kernel(
    const _Float16* __restrict__ A, const _Float16* __restrict__ W,
    const float* __restrict__ bias, void* __restrict__ outp,
    int M, int N, int K, int mode) {
  int tiles_n = (N + 15) >> 4;
  int tiles_n4 = (tiles_n + 3) >> 2;
  int tiles_m = M >> 4;
  int wid = blockIdx.x * 4 + (threadIdx.x >> 5);
  if (wid >= tiles_m * tiles_n4) return;
  int tm = wid / tiles_n4, tn0 = (wid % tiles_n4) * 4;
  int l = threadIdx.x & 31, half_sel = l >> 4, ln = l & 15;
  const _Float16* Ab = A + (size_t)(tm * 16 + ln) * K;
  const _Float16* Wb[4];
#pragma unroll
  for (int j = 0; j < 4; j++) {
    int tnj = tn0 + j; if (tnj > tiles_n - 1) tnj = tiles_n - 1; // clamp into padded region
    Wb[j] = W + (size_t)(tnj * 16 + ln) * K;
  }
  v8f acc[4];
#pragma unroll
  for (int j = 0; j < 4; j++) acc[j] = v8f{0.f, 0.f, 0.f, 0.f, 0.f, 0.f, 0.f, 0.f};
  for (int kk = 0; kk < K; kk += 32) {
    V16 a;
    a.u4[0] = *(const uint4*)(Ab + kk + half_sel * 16);
    a.u4[1] = *(const uint4*)(Ab + kk + half_sel * 16 + 8);
    __builtin_prefetch(Ab + kk + 64, 0, 1);
#pragma unroll
    for (int j = 0; j < 4; j++) {
      V16 b;
      b.u4[0] = *(const uint4*)(Wb[j] + kk + half_sel * 16);
      b.u4[1] = *(const uint4*)(Wb[j] + kk + half_sel * 16 + 8);
      acc[j] = wmma_f16(a.v, b.v, acc[j]);
    }
  }
#pragma unroll
  for (int j = 0; j < 4; j++) {
    int tnj = tn0 + j;
    if (tnj >= tiles_n) continue;
#pragma unroll
    for (int r = 0; r < 8; r++) {
      int m = tm * 16 + r + 8 * half_sel;
      int n = tnj * 16 + ln;
      if (n < N) {
        float v = acc[j][r] + (bias ? bias[n] : 0.0f);
        if (mode == 0)       ((_Float16*)outp)[(size_t)m * N + n] = (_Float16)v;
        else if (mode == 1)  ((_Float16*)outp)[(size_t)m * N + n] = (_Float16)geluf(v);
        else if (mode == 2)  ((float*)outp)[(size_t)m * N + n] += v;
        else                 ((float*)outp)[(size_t)m * N + n] = v;
      }
    }
  }
}

// ---------------- fused attention per (b, h) ----------------
__global__ __launch_bounds__(128) void attn_kernel(const _Float16* __restrict__ qkv,
                                                   const int* __restrict__ mask,
                                                   _Float16* __restrict__ att) {
  __shared__ _Float16 qs[S_ * 16], ks[S_ * 16], vs[S_ * 16];
  __shared__ _Float16 sc[S_ * S_];
  __shared__ int km[S_];
  int b = blockIdx.x >> 3, h = blockIdx.x & 7;
  int t = threadIdx.x;
  const _Float16* row = qkv + (size_t)(b * S_ + t) * 384;
  for (int d = 0; d < 16; d++) {
    qs[t * 16 + d] = row[h * 16 + d];
    ks[t * 16 + d] = row[128 + h * 16 + d];
    vs[t * 16 + d] = row[256 + h * 16 + d];
  }
  km[t] = mask[b * S_ + t];
  __syncthreads();
  int w = t >> 5, l = t & 31, half_sel = l >> 4, ln = l & 15;
  // scores: 8x8 tiles of 16x16, K=16 zero-padded to 32
  for (int tile = w; tile < 64; tile += 4) {
    int ti = tile >> 3, tj = tile & 7;
    V16 a, bv;
    for (int e = 0; e < 16; e++) {
      int ka = half_sel * 16 + e;
      a.h[e] = (ka < 16) ? qs[(ti * 16 + ln) * 16 + ka] : (_Float16)0.0f;
      int kb = half_sel * 16 + e;
      bv.h[e] = (kb < 16) ? ks[(tj * 16 + ln) * 16 + kb] : (_Float16)0.0f;
    }
    v8f acc = {0.f, 0.f, 0.f, 0.f, 0.f, 0.f, 0.f, 0.f};
    acc = wmma_f16(a.v, bv.v, acc);
    for (int r = 0; r < 8; r++) {
      int rowi = ti * 16 + r + 8 * half_sel;
      int col = tj * 16 + ln;
      float sv = acc[r] * 0.25f;          // 1/sqrt(DH)
      if (!km[col]) sv = -1e9f;           // -> -inf in f16, exp()=0
      sc[rowi * S_ + col] = (_Float16)sv;
    }
  }
  __syncthreads();
  // softmax: thread t owns row t
  float mx = -1e30f;
  for (int c = 0; c < S_; c++) mx = fmaxf(mx, (float)sc[t * S_ + c]);
  float sm = 0.0f;
  for (int c = 0; c < S_; c++) sm += expf((float)sc[t * S_ + c] - mx);
  float inv = 1.0f / sm;
  for (int c = 0; c < S_; c++)
    sc[t * S_ + c] = (_Float16)(expf((float)sc[t * S_ + c] - mx) * inv);
  __syncthreads();
  // P @ V : 8 m-tiles x 1 n-tile, K=128 in 4 WMMA steps
  for (int mt = w; mt < 8; mt += 4) {
    v8f acc = {0.f, 0.f, 0.f, 0.f, 0.f, 0.f, 0.f, 0.f};
    for (int kk = 0; kk < 4; kk++) {
      V16 a, bv;
      for (int e = 0; e < 16; e++) {
        int ka = kk * 32 + half_sel * 16 + e;
        a.h[e] = sc[(mt * 16 + ln) * S_ + ka];
        int kb = kk * 32 + half_sel * 16 + e;
        bv.h[e] = vs[kb * 16 + ln];
      }
      acc = wmma_f16(a.v, bv.v, acc);
    }
    for (int r = 0; r < 8; r++) {
      int srow = mt * 16 + r + 8 * half_sel;
      att[(size_t)(b * S_ + srow) * D_ + h * 16 + ln] = (_Float16)acc[r];
    }
  }
}

// ---------------- gather last valid token -> f16 ----------------
__global__ void last_kernel(const float* __restrict__ x, const int* __restrict__ vlen,
                            _Float16* __restrict__ last16) {
  int b = blockIdx.x, j = threadIdx.x;
  int li = vlen[b] - 1;
  last16[b * D_ + j] = (_Float16)x[((size_t)b * S_ + li) * D_ + j];
}

// ---------------- fused scores GEMM (B x L x K=64) + ensemble epilogue ----------------
// Each wave: TDM-loads 64 loc-embedding rows (8KB) into its LDS chunk, then
// computes a 16x64 output strip with 8 WMMAs fed from LDS, then applies the
// hist/top-k/visited ensemble epilogue.
__global__ __launch_bounds__(128) void final_kernel(
    const _Float16* __restrict__ q16, const _Float16* __restrict__ loc16,
    const float* __restrict__ hist, const float* __restrict__ rowmax,
    const int* __restrict__ topk_pos, const float* __restrict__ dense,
    const unsigned* __restrict__ visited, const float* __restrict__ ens_alpha,
    float* __restrict__ out) {
  __shared__ _Float16 lbuf[4][64 * 64];   // 32 KB: 4 waves x (64 rows x 64 cols)
  int w = threadIdx.x >> 5, l = threadIdx.x & 31, half_sel = l >> 4, ln = l & 15;
  const int ngroups = (L_ / 16) / 4;      // 625 groups of 4 n-tiles
  int wid = blockIdx.x * 4 + w;           // exactly 16*625 waves in grid
  int tm = wid / ngroups, ng = wid % ngroups;
  int row0 = ng * 64;

  // ---- Tensor Data Mover: 2D tile 64(cols) x 64(rows), 2B elements ----
  unsigned lds_off = (unsigned)(unsigned long long)(uintptr_t)&lbuf[w][0];
  unsigned long long ga = (unsigned long long)(uintptr_t)(loc16 + (size_t)row0 * 64);
  v4u g0;
  g0[0] = 1u;                                              // count=1 valid descriptor
  g0[1] = lds_off;                                         // lds_addr
  g0[2] = (unsigned)(ga & 0xFFFFFFFFu);                    // global_addr[31:0]
  g0[3] = (unsigned)((ga >> 32) & 0x1FFFFFFu) | (2u << 30);// global_addr[56:32] | type=2
  v8i_t g1;
  g1[0] = 0x00010000;                 // data_size=1 (2 bytes)
  g1[1] = (64 & 0xFFFF) << 16;        // tensor_dim0 = 64
  g1[2] = (64 & 0xFFFF) << 16;        // tensor_dim0 hi=0 | tensor_dim1 = 64
  g1[3] = (64 & 0xFFFF) << 16;        // tensor_dim1 hi=0 | tile_dim0 = 64
  g1[4] = (64 & 0xFFFF);              // tile_dim1 = 64, tile_dim2 = 0
  g1[5] = 64;                         // tensor_dim0_stride = 64
  g1[6] = 0; g1[7] = 0;
  v4i_t gz4 = {0, 0, 0, 0};
  v8i_t gz8 = {0, 0, 0, 0, 0, 0, 0, 0};
  __builtin_amdgcn_tensor_load_to_lds(g0, g1, gz4, gz4, gz8, 0);
  __builtin_amdgcn_s_wait_tensorcnt(0);

  const _Float16* Ab = q16 + (size_t)(tm * 16 + ln) * 64;
  v8f acc[4];
#pragma unroll
  for (int j = 0; j < 4; j++) acc[j] = v8f{0.f, 0.f, 0.f, 0.f, 0.f, 0.f, 0.f, 0.f};
#pragma unroll
  for (int kk = 0; kk < 64; kk += 32) {
    V16 a;
    a.u4[0] = *(const uint4*)(Ab + kk + half_sel * 16);
    a.u4[1] = *(const uint4*)(Ab + kk + half_sel * 16 + 8);
#pragma unroll
    for (int j = 0; j < 4; j++) {
      V16 b;
      const _Float16* Bp = &lbuf[w][(j * 16 + ln) * 64 + kk + half_sel * 16];
      b.u4[0] = *(const uint4*)Bp;
      b.u4[1] = *(const uint4*)(Bp + 8);
      acc[j] = wmma_f16(a.v, b.v, acc[j]);
    }
  }
  float alpha = sigmoidf_(ens_alpha[0]);
#pragma unroll
  for (int j = 0; j < 4; j++) {
#pragma unroll
    for (int r = 0; r < 8; r++) {
      int bidx = tm * 16 + r + 8 * half_sel;
      int loc = row0 + j * 16 + ln;
      float score = acc[j][r];
      float learned = -20.0f;
      int tp = topk_pos[loc];
      if (tp >= 0) learned = dense[(size_t)bidx * TOPK_ + tp];
      if ((visited[bidx * VISW_ + (loc >> 5)] >> (loc & 31)) & 1u)
        learned = fmaxf(learned, score);
      float mxv = rowmax[bidx];
      float hn = hist[(size_t)bidx * L_ + loc] / ((mxv > 0.0f) ? mxv : 1.0f) * 10.0f;
      out[(size_t)bidx * L_ + loc] = alpha * hn + (1.0f - alpha) * learned;
    }
  }
}

// ================= host side =================
static inline void conv_w(const float* src, _Float16* dst, int n, hipStream_t s) {
  hipLaunchKernelGGL(f32_to_f16_kernel, dim3((n + 255) / 256), dim3(256), 0, s, src, dst, n);
}

extern "C" void kernel_launch(void* const* d_in, const int* in_sizes, int n_in,
                              void* d_out, int out_size, void* d_ws, size_t ws_size,
                              hipStream_t stream) {
  const int*   loc_seq    = (const int*)d_in[0];
  const int*   user_seq   = (const int*)d_in[1];
  const int*   weekday    = (const int*)d_in[2];
  const int*   start_min  = (const int*)d_in[3];
  const int*   dur        = (const int*)d_in[4];
  const int*   diff       = (const int*)d_in[5];
  const int*   mask       = (const int*)d_in[6];
  const int*   topk_idx   = (const int*)d_in[7];
  const float* loc_freq   = (const float*)d_in[8];
  const float* loc_emb    = (const float*)d_in[9];
  const float* user_emb   = (const float*)d_in[10];
  const float* tproj_w    = (const float*)d_in[11];
  const float* tproj_b    = (const float*)d_in[12];
  const float* tln_g      = (const float*)d_in[13];
  const float* tln_b      = (const float*)d_in[14];
  const float* in_ln_g    = (const float*)d_in[15];
  const float* in_ln_b    = (const float*)d_in[16];
  const float* Wqkv       = (const float*)d_in[17];
  const float* bqkv       = (const float*)d_in[18];
  const float* Wo         = (const float*)d_in[19];
  const float* bo         = (const float*)d_in[20];
  const float* ln1_g      = (const float*)d_in[21];
  const float* ln1_b      = (const float*)d_in[22];
  const float* lin1_w     = (const float*)d_in[23];
  const float* lin1_b     = (const float*)d_in[24];
  const float* lin2_w     = (const float*)d_in[25];
  const float* lin2_b     = (const float*)d_in[26];
  const float* ln2_g      = (const float*)d_in[27];
  const float* ln2_b      = (const float*)d_in[28];
  const float* dp1_w      = (const float*)d_in[29];
  const float* dp1_b      = (const float*)d_in[30];
  const float* dp2_w      = (const float*)d_in[31];
  const float* dp2_b      = (const float*)d_in[32];
  const float* cp_w       = (const float*)d_in[33];
  const float* cp_b       = (const float*)d_in[34];
  const float* cln_g      = (const float*)d_in[35];
  const float* cln_b      = (const float*)d_in[36];
  const float* hd         = (const float*)d_in[37];
  const float* rw         = (const float*)d_in[38];
  const float* ens_alpha  = (const float*)d_in[39];
  float* out = (float*)d_out;

  // ---- workspace carve-up (256B aligned regions) ----
  char* ws = (char*)d_ws;
  size_t off = 0;
  auto carve = [&](size_t bytes) -> char* {
    char* p = ws + off;
    off = (off + bytes + 255) & ~(size_t)255;
    return p;
  };
  const int DP2P = 2512;   // 2500 padded to multiple of 16 rows
  const int CPP  = 64;     // 56 padded to multiple of 16 rows
  float*     hist     = (float*)carve((size_t)B_ * L_ * 4);
  unsigned*  visited  = (unsigned*)carve((size_t)B_ * VISW_ * 4);
  int*       vlen_ws  = (int*)carve(B_ * 4);
  float*     rowmax   = (float*)carve(B_ * 4);
  int*       topk_pos = (int*)carve(L_ * 4);
  float*     x        = (float*)carve((size_t)B_ * S_ * D_ * 4);
  _Float16*  h16      = (_Float16*)carve((size_t)B_ * S_ * D_ * 2);
  _Float16*  qkv16    = (_Float16*)carve((size_t)B_ * S_ * 3 * D_ * 2);
  _Float16*  att16    = (_Float16*)carve((size_t)B_ * S_ * D_ * 2);
  _Float16*  mid16    = (_Float16*)carve((size_t)B_ * S_ * DFF_ * 2);
  _Float16*  last16   = (_Float16*)carve(B_ * D_ * 2);
  _Float16*  mid2_16  = (_Float16*)carve(B_ * DFF_ * 2);
  float*     dense    = (float*)carve((size_t)B_ * TOPK_ * 4);
  float*     qpre     = (float*)carve(B_ * 56 * 4);
  _Float16*  q16      = (_Float16*)carve(B_ * 64 * 2);
  _Float16*  loc16    = (_Float16*)carve((size_t)L_ * 64 * 2);
  _Float16*  Wqkv16   = (_Float16*)carve((size_t)NL_ * 3 * D_ * D_ * 2);
  _Float16*  Wo16     = (_Float16*)carve((size_t)NL_ * D_ * D_ * 2);
  _Float16*  lin1_16  = (_Float16*)carve((size_t)NL_ * DFF_ * D_ * 2);
  _Float16*  lin2_16  = (_Float16*)carve((size_t)NL_ * D_ * DFF_ * 2);
  _Float16*  dp1_16   = (_Float16*)carve((size_t)DFF_ * D_ * 2);
  _Float16*  dp2_16   = (_Float16*)carve((size_t)DP2P * DFF_ * 2);   // zero-padded rows
  _Float16*  cp_16    = (_Float16*)carve((size_t)CPP * D_ * 2);      // zero-padded rows
  (void)ws_size; (void)n_in; (void)in_sizes; (void)out_size;

  // ---- init + weight conversion ----
  (void)hipMemsetAsync(hist, 0, (size_t)B_ * L_ * 4, stream);
  (void)hipMemsetAsync(visited, 0, (size_t)B_ * VISW_ * 4, stream);
  (void)hipMemsetAsync(topk_pos, 0xFF, (size_t)L_ * 4, stream);      // -1
  (void)hipMemsetAsync(dp2_16, 0, (size_t)DP2P * DFF_ * 2, stream);  // zero pad rows
  (void)hipMemsetAsync(cp_16, 0, (size_t)CPP * D_ * 2, stream);
  conv_w(Wqkv,   Wqkv16,  NL_ * 3 * D_ * D_, stream);
  conv_w(Wo,     Wo16,    NL_ * D_ * D_,     stream);
  conv_w(lin1_w, lin1_16, NL_ * DFF_ * D_,   stream);
  conv_w(lin2_w, lin2_16, NL_ * D_ * DFF_,   stream);
  conv_w(dp1_w,  dp1_16,  DFF_ * D_,         stream);
  conv_w(dp2_w,  dp2_16,  TOPK_ * DFF_,      stream);
  conv_w(cp_w,   cp_16,   56 * D_,           stream);
  hipLaunchKernelGGL(locpad_kernel, dim3((L_ * 64 + 255) / 256), dim3(256), 0, stream, loc_emb, loc16);
  hipLaunchKernelGGL(topk_scatter_kernel, dim3((TOPK_ + 255) / 256), dim3(256), 0, stream, topk_idx, topk_pos);

  // ---- history ----
  hipLaunchKernelGGL(hist_kernel, dim3(B_), dim3(S_), 0, stream,
                     loc_seq, mask, hd, rw, hist, visited, vlen_ws);
  hipLaunchKernelGGL(histmax_kernel, dim3(B_), dim3(256), 0, stream, loc_freq, hist, rowmax);

  // ---- embedding / input ----
  hipLaunchKernelGGL(embed_kernel, dim3(B_ * S_), dim3(D_), 0, stream,
                     loc_seq, user_seq, weekday, start_min, dur, diff,
                     loc_emb, user_emb, tproj_w, tproj_b, tln_g, tln_b,
                     in_ln_g, in_ln_b, x);

  // ---- transformer layers ----
  const int M = B_ * S_;
  auto gemm = [&](const _Float16* A, const _Float16* W, const float* bias, void* o,
                  int m, int n, int k, int mode) {
    int tiles_n4 = (((n + 15) / 16) + 3) / 4;
    int waves = (m / 16) * tiles_n4;
    hipLaunchKernelGGL(gemm16_kernel, dim3((waves + 3) / 4), dim3(128), 0, stream,
                       A, W, bias, o, m, n, k, mode);
  };
  for (int l = 0; l < NL_; l++) {
    hipLaunchKernelGGL(ln_kernel, dim3(M), dim3(128), 0, stream,
                       x, ln1_g + l * D_, ln1_b + l * D_, h16, D_, D_);
    gemm(h16, Wqkv16 + (size_t)l * 3 * D_ * D_, bqkv + l * 3 * D_, qkv16, M, 3 * D_, D_, 0);
    hipLaunchKernelGGL(attn_kernel, dim3(B_ * NH_), dim3(128), 0, stream, qkv16, mask, att16);
    gemm(att16, Wo16 + (size_t)l * D_ * D_, bo + l * D_, x, M, D_, D_, 2);
    hipLaunchKernelGGL(ln_kernel, dim3(M), dim3(128), 0, stream,
                       x, ln2_g + l * D_, ln2_b + l * D_, h16, D_, D_);
    gemm(h16, lin1_16 + (size_t)l * DFF_ * D_, lin1_b + l * DFF_, mid16, M, DFF_, D_, 1);
    gemm(mid16, lin2_16 + (size_t)l * D_ * DFF_, lin2_b + l * D_, x, M, D_, DFF_, 2);
  }

  // ---- heads ----
  hipLaunchKernelGGL(last_kernel, dim3(B_), dim3(D_), 0, stream, x, vlen_ws, last16);
  gemm(last16, dp1_16, dp1_b, mid2_16, B_, DFF_, D_, 1);          // gelu f16
  gemm(mid2_16, dp2_16, dp2_b, dense, B_, TOPK_, DFF_, 3);        // f32
  gemm(last16, cp_16, cp_b, qpre, B_, 56, D_, 3);                 // f32
  hipLaunchKernelGGL(ln_kernel, dim3(B_), dim3(64), 0, stream, qpre, cln_g, cln_b, q16, 56, 64);

  // ---- fused scores + ensemble (TDM + WMMA) ----
  hipLaunchKernelGGL(final_kernel, dim3((B_ / 16) * ((L_ / 16) / 4) / 4), dim3(128), 0, stream,
                     q16, loc16, hist, rowmax, topk_pos, dense, visited, ens_alpha, out);
}
